// ChannelIndependentAttention_43885975830990
// MI455X (gfx1250) — matrix operand order repaired and checked by reading, not compiled, not run on hardware
//
#include <hip/hip_runtime.h>

// ---------------------------------------------------------------------------
// Types for CDNA5 WMMA (wave32): v_wmma_f32_16x16x32_bf16
// ---------------------------------------------------------------------------
typedef __attribute__((ext_vector_type(16))) __bf16 v16bf;
typedef __attribute__((ext_vector_type(8)))  __bf16 v8bf;
typedef __attribute__((ext_vector_type(8)))  float  v8f;
typedef unsigned int u32x4 __attribute__((ext_vector_type(4)));
typedef int          i32x8 __attribute__((ext_vector_type(8)));
typedef int          i32x4 __attribute__((ext_vector_type(4)));

#define DMODEL 1024
#define NH     16
#define DK     64
#define BATCH  4
#define SEQ    2048
#define ROWS   (BATCH * SEQ)   // 8192
#define NQKV   (3 * DMODEL)    // 3072

#if __has_builtin(__builtin_amdgcn_tensor_load_to_lds) && \
    __has_builtin(__builtin_amdgcn_s_wait_tensorcnt)
#define USE_TDM 1
#else
#define USE_TDM 0
#endif

// A-fragment (16x32 bf16, MxK): lane L holds row M = L&15.
//   lanes 0-15 : K = 0..15
//   lanes 16-31: K = 16..31
static __device__ inline v16bf make_afrag(const __bf16* rowptr, int lane) {
    const int kb = (lane >> 4) << 4;                       // 0 or 16
    return *(const v16bf*)(rowptr + kb);
}

// B-fragment (32x16 bf16, KxN) from an LDS tile stored TRANSPOSED as [n][k]:
//   lane L holds column N = L&15; lanes 0-15 hold K=0..15, lanes 16-31 K=16..31.
static __device__ inline v16bf make_bfrag(const __bf16* tileT, int stride,
                                          int nbase, int kbase, int lane) {
    const int n = nbase + (lane & 15);
    const int k = kbase + ((lane >> 4) << 4);              // +0 or +16
    return *(const v16bf*)(tileT + (size_t)n * stride + k);
}

static __device__ inline v8f wmma_bf16(v16bf a, v16bf b, v8f c) {
    return __builtin_amdgcn_wmma_f32_16x16x32_bf16(false, a, false, b,
                                                   (short)0, c, false, false);
}

#if USE_TDM
// Tensor Data Mover: 2-D tile (bf16 elements) global -> LDS.
// D# packing per CDNA5 ISA.
// This toolchain exposes the 6-arg builtin:
//   (uint32x4 g0, int32x8 g1, int32x4 g2, int32x4 g3, int32x8, i32 cpol)
static __device__ inline void tdm_load_2d(unsigned lds_addr, const void* gptr,
                                          unsigned tile0, unsigned tile1,
                                          unsigned tdim0, unsigned tdim1,
                                          unsigned stride0 /* elements */) {
    const unsigned long long ga = (unsigned long long)(uintptr_t)gptr;
    u32x4 g0;
    g0[0] = 1u;                                            // count=1, user mode
    g0[1] = lds_addr;                                      // LDS byte address
    g0[2] = (unsigned)(ga & 0xffffffffu);                  // global_addr[31:0]
    g0[3] = (unsigned)((ga >> 32) & 0x01ffffffu)           // global_addr[56:32]
          | 0x80000000u;                                   // type=2 ("image")
    i32x8 g1;
    g1[0] = (int)(1u << 16);                               // data_size=1 -> 2 bytes
    g1[1] = (int)(tdim0 << 16);                            // tensor_dim0[15:0]
    g1[2] = (int)((tdim0 >> 16) | (tdim1 << 16));          // tdim0[31:16] | tdim1[15:0]
    g1[3] = (int)((tdim1 >> 16) | (tile0 << 16));          // tdim1[31:16] | tile_dim0
    g1[4] = (int)(tile1 & 0xffffu);                        // tile_dim1 (tile_dim2=0)
    g1[5] = (int)stride0;                                  // tensor_dim0_stride[31:0]
    g1[6] = 0;                                             // stride0[47:32] | dim1_stride lo
    g1[7] = 0;
    const i32x4 z4 = {0, 0, 0, 0};                         // groups 2/3 unused (2-D)
    const i32x8 z8 = {0, 0, 0, 0, 0, 0, 0, 0};
    __builtin_amdgcn_tensor_load_to_lds(g0, g1, z4, z4, z8, 0);
}
#endif

// ---------------------------------------------------------------------------
// Kernel 1: qkv = x @ W + b, fp32 in -> bf16 out, scattered per head.
//   q  : [B*H][T][dk]  (pre-scaled by 1/sqrt(dk)),  k : [B*H][T][dk],
//   vT : [B*H][dk][T]  (transposed so PV B-fragments are contiguous)
// ---------------------------------------------------------------------------
__global__ __launch_bounds__(256)
void qkv_gemm_kernel(const float* __restrict__ x, const float* __restrict__ W,
                     const float* __restrict__ bias,
                     __bf16* __restrict__ q, __bf16* __restrict__ k,
                     __bf16* __restrict__ vT)
{
    __shared__ __bf16 As[128][32];   // x tile, row-major [m][k]
    __shared__ __bf16 Bs[128][32];   // W tile, TRANSPOSED [n][k]

    const int t    = threadIdx.x;
    const int lane = t & 31;
    const int w    = t >> 5;
    const int wm   = w >> 2;         // 0..1 -> 64 rows each
    const int wn   = w & 3;          // 0..3 -> 32 cols each
    const int n0   = blockIdx.x * 128;
    const int m0   = blockIdx.y * 128;

    v8f acc[4][2];
    #pragma unroll
    for (int i = 0; i < 4; ++i)
        #pragma unroll
        for (int j = 0; j < 2; ++j)
            acc[i][j] = {};

    for (int k0 = 0; k0 < DMODEL; k0 += 32) {
        __syncthreads();
        #pragma unroll
        for (int p = 0; p < 4; ++p) {
            const int row = p * 32 + (t >> 3);
            const int cc  = (t & 7) * 4;
            const float4 v = *(const float4*)(x + (size_t)(m0 + row) * DMODEL + k0 + cc);
            As[row][cc + 0] = (__bf16)v.x;
            As[row][cc + 1] = (__bf16)v.y;
            As[row][cc + 2] = (__bf16)v.z;
            As[row][cc + 3] = (__bf16)v.w;
        }
        #pragma unroll
        for (int p = 0; p < 4; ++p) {
            const int kk = p * 8 + (t >> 5);
            const int nb = (t & 31) * 4;
            const float4 v = *(const float4*)(W + (size_t)(k0 + kk) * NQKV + n0 + nb);
            Bs[nb + 0][kk] = (__bf16)v.x;
            Bs[nb + 1][kk] = (__bf16)v.y;
            Bs[nb + 2][kk] = (__bf16)v.z;
            Bs[nb + 3][kk] = (__bf16)v.w;
        }
        __syncthreads();

        v16bf af[4], bf2[2];
        #pragma unroll
        for (int i = 0; i < 4; ++i)
            af[i] = make_afrag(&As[wm * 64 + i * 16 + (lane & 15)][0], lane);
        #pragma unroll
        for (int j = 0; j < 2; ++j)
            bf2[j] = make_bfrag(&Bs[0][0], 32, wn * 32 + j * 16, 0, lane);

        #pragma unroll
        for (int i = 0; i < 4; ++i)
            #pragma unroll
            for (int j = 0; j < 2; ++j)
                acc[i][j] = wmma_bf16(af[i], bf2[j], acc[i][j]);
    }

    #pragma unroll
    for (int i = 0; i < 4; ++i) {
        #pragma unroll
        for (int j = 0; j < 2; ++j) {
            const int colg  = n0 + wn * 32 + j * 16 + (lane & 15);
            const float bb  = bias[colg];
            const int which = colg >> 10;       // 0=q 1=k 2=v
            const int rem   = colg & 1023;
            const int hh    = rem >> 6;
            const int dd    = rem & 63;
            #pragma unroll
            for (int r = 0; r < 8; ++r) {
                const int rowg = m0 + wm * 64 + i * 16 + r + ((lane >> 4) << 3);
                const int bidx = rowg >> 11;    // /SEQ
                const int tt   = rowg & 2047;
                const int bh   = bidx * NH + hh;
                const float v  = acc[i][j][r] + bb;
                if (which == 0)
                    q[((size_t)bh * SEQ + tt) * DK + dd] = (__bf16)(v * 0.125f);
                else if (which == 1)
                    k[((size_t)bh * SEQ + tt) * DK + dd] = (__bf16)v;
                else
                    vT[((size_t)bh * DK + dd) * SEQ + tt] = (__bf16)v;
            }
        }
    }
}

// ---------------------------------------------------------------------------
// Kernel 2: flash attention, TDM-double-buffered KV streaming.
// Block = 128 threads (4 waves), each wave owns 16 query rows.
// ---------------------------------------------------------------------------
__global__ __launch_bounds__(128)
void flash_attn_kernel(const __bf16* __restrict__ q, const __bf16* __restrict__ kmat,
                       const __bf16* __restrict__ vT, float* __restrict__ out)
{
    __shared__ __bf16 Kt[2][64][64];    // keys  [key][d], double-buffered
    __shared__ __bf16 Vt[2][64][64];    // v^T   [d][key], double-buffered
    __shared__ __bf16 Pt[4][16][64];    // per-wave softmax tile (wave-private)

    const int t    = threadIdx.x;
    const int lane = t & 31;
    const int w    = t >> 5;
    const int bh   = blockIdx.y;
    const int b    = bh >> 4;
    const int h    = bh & 15;
    const int q0   = blockIdx.x * 64 + w * 16;

    const __bf16* qbase = q    + (size_t)bh * SEQ * DK;
    const __bf16* kbase = kmat + (size_t)bh * SEQ * DK;
    const __bf16* vbase = vT   + (size_t)bh * DK * SEQ;

    // Issue (or perform) the tile load for kv block starting at j0 into buffer buf.
    auto issue_tile = [&](int buf, int j0) {
#if USE_TDM
        if (w == 0) {   // TDM is wave-level DMA; one issuing wave is enough
            tdm_load_2d((unsigned)(uintptr_t)&Kt[buf][0][0],
                        kbase + (size_t)j0 * DK,
                        /*tile*/ 64, 64, /*tensor*/ DK, SEQ, /*stride0*/ DK);
            tdm_load_2d((unsigned)(uintptr_t)&Vt[buf][0][0],
                        vbase + j0,
                        /*tile*/ 64, 64, /*tensor*/ SEQ, DK, /*stride0*/ SEQ);
        }
#else
        const int row = t >> 1;
        const int col = (t & 1) * 32;
        const uint4* sk = (const uint4*)(kbase + (size_t)(j0 + row) * DK + col);
        uint4* dk = (uint4*)(&Kt[buf][row][col]);
        dk[0] = sk[0]; dk[1] = sk[1]; dk[2] = sk[2]; dk[3] = sk[3];
        const uint4* sv = (const uint4*)(vbase + (size_t)row * SEQ + j0 + col);
        uint4* dv = (uint4*)(&Vt[buf][row][col]);
        dv[0] = sv[0]; dv[1] = sv[1]; dv[2] = sv[2]; dv[3] = sv[3];
#endif
    };

    // Persistent q fragments (A-matrix, K-dim = d in two 32-chunks)
    const __bf16* qrow = qbase + (size_t)(q0 + (lane & 15)) * DK;
    const v16bf aq0 = make_afrag(qrow +  0, lane);
    const v16bf aq1 = make_afrag(qrow + 32, lane);

    v8f acc[4];
    #pragma unroll
    for (int i = 0; i < 4; ++i) acc[i] = {};
    float mstate[8], lstate[8];
    #pragma unroll
    for (int r = 0; r < 8; ++r) { mstate[r] = -1e30f; lstate[r] = 0.0f; }

    constexpr int NT = SEQ / 64;      // 32 kv tiles
    issue_tile(0, 0);                 // prologue: prefetch tile 0

    for (int it = 0; it < NT; ++it) {
        const int buf = it & 1;
        if (it + 1 < NT) {
            issue_tile(buf ^ 1, (it + 1) * 64);   // prefetch next tile
#if USE_TDM
            __builtin_amdgcn_s_wait_tensorcnt(2); // oldest pair (current tile) done
        } else {
            __builtin_amdgcn_s_wait_tensorcnt(0);
#endif
        }
        __syncthreads();              // current tile visible to all waves

        // ---- scores: s = q @ k^T (16 queries x 64 keys) ----
        v8f s[4];
        #pragma unroll
        for (int nt = 0; nt < 4; ++nt) {
            s[nt] = {};
            const v16bf b0 = make_bfrag(&Kt[buf][0][0], 64, nt * 16,  0, lane);
            const v16bf b1 = make_bfrag(&Kt[buf][0][0], 64, nt * 16, 32, lane);
            s[nt] = wmma_bf16(aq0, b0, s[nt]);
            s[nt] = wmma_bf16(aq1, b1, s[nt]);
        }

        // ---- online softmax (row reductions across 16-lane N groups) ----
        #pragma unroll
        for (int r = 0; r < 8; ++r) {
            float mx = fmaxf(fmaxf(s[0][r], s[1][r]), fmaxf(s[2][r], s[3][r]));
            mx = fmaxf(mx, __shfl_xor(mx, 1, 32));
            mx = fmaxf(mx, __shfl_xor(mx, 2, 32));
            mx = fmaxf(mx, __shfl_xor(mx, 4, 32));
            mx = fmaxf(mx, __shfl_xor(mx, 8, 32));
            const float mn = fmaxf(mstate[r], mx);
            const float sc = __expf(mstate[r] - mn);
            const float p0 = __expf(s[0][r] - mn);
            const float p1 = __expf(s[1][r] - mn);
            const float p2 = __expf(s[2][r] - mn);
            const float p3 = __expf(s[3][r] - mn);
            float rs = p0 + p1 + p2 + p3;
            rs += __shfl_xor(rs, 1, 32);
            rs += __shfl_xor(rs, 2, 32);
            rs += __shfl_xor(rs, 4, 32);
            rs += __shfl_xor(rs, 8, 32);
            lstate[r] = lstate[r] * sc + rs;
            mstate[r] = mn;
            acc[0][r] *= sc; acc[1][r] *= sc; acc[2][r] *= sc; acc[3][r] *= sc;
            // spill P to wave-private LDS (C layout -> row-major bf16)
            const int mrow = r + ((lane >> 4) << 3);
            const int nc   = lane & 15;
            Pt[w][mrow][ 0 + nc] = (__bf16)p0;
            Pt[w][mrow][16 + nc] = (__bf16)p1;
            Pt[w][mrow][32 + nc] = (__bf16)p2;
            Pt[w][mrow][48 + nc] = (__bf16)p3;
        }
        // Pt[w] is wave-private: compiler-inserted s_wait_dscnt orders it.

        // ---- acc += P @ V ----
        const __bf16* prow = &Pt[w][lane & 15][0];
        const v16bf ap0 = make_afrag(prow +  0, lane);
        const v16bf ap1 = make_afrag(prow + 32, lane);
        #pragma unroll
        for (int nt = 0; nt < 4; ++nt) {
            const v16bf b0 = make_bfrag(&Vt[buf][0][0], 64, nt * 16,  0, lane);
            const v16bf b1 = make_bfrag(&Vt[buf][0][0], 64, nt * 16, 32, lane);
            acc[nt] = wmma_bf16(ap0, b0, acc[nt]);
            acc[nt] = wmma_bf16(ap1, b1, acc[nt]);
        }
        __syncthreads();              // everyone done with buf before overwrite
    }

    // ---- epilogue: normalize, store fp32 (B, T, H*dk) ----
    #pragma unroll
    for (int r = 0; r < 8; ++r) {
        const float inv  = 1.0f / lstate[r];
        const int   mrow = r + ((lane >> 4) << 3);
        const int   trow = q0 + mrow;
        const size_t base = ((size_t)b * SEQ + trow) * DMODEL + h * DK;
        #pragma unroll
        for (int nt = 0; nt < 4; ++nt)
            out[base + nt * 16 + (lane & 15)] = acc[nt][r] * inv;
    }
}

// ---------------------------------------------------------------------------
extern "C" void kernel_launch(void* const* d_in, const int* in_sizes, int n_in,
                              void* d_out, int out_size, void* d_ws, size_t ws_size,
                              hipStream_t stream)
{
    (void)in_sizes; (void)n_in; (void)out_size; (void)ws_size;
    const float* x    = (const float*)d_in[0];
    const float* W    = (const float*)d_in[1];
    const float* bias = (const float*)d_in[2];
    float* out        = (float*)d_out;

    char* ws = (char*)d_ws;
    __bf16* qws = (__bf16*)(ws);                                  // 16 MB
    __bf16* kws = (__bf16*)(ws + (size_t)ROWS * DMODEL * 2);      // 16 MB
    __bf16* vws = (__bf16*)(ws + (size_t)ROWS * DMODEL * 4);      // 16 MB (v^T)

    dim3 gGemm(NQKV / 128, ROWS / 128);   // 24 x 64
    qkv_gemm_kernel<<<gGemm, 256, 0, stream>>>(x, W, bias, qws, kws, vws);

    dim3 gAttn(SEQ / 64, BATCH * NH);     // 32 x 64
    flash_attn_kernel<<<gAttn, 128, 0, stream>>>(qws, kws, vws, out);
}
